// GINet_2D_7859790152037
// MI455X (gfx1250) — matrix operand order repaired and checked by reading, not compiled, not run on hardware
//
#include <hip/hip_runtime.h>

#define DIM 256
#define NN 20000
#define EE 320000
#define GG 256
#define LL 5

typedef __attribute__((ext_vector_type(16))) __bf16 v16bf;
typedef __attribute__((ext_vector_type(8)))  float  v8f;

__device__ __forceinline__ unsigned short f2bf(float f) {
    unsigned int u = __float_as_uint(f);
    u += 0x7FFFu + ((u >> 16) & 1u);      // round-to-nearest-even
    return (unsigned short)(u >> 16);
}

__device__ __forceinline__ unsigned int pk2bf(float a, float b) {
    return (unsigned int)f2bf(a) | ((unsigned int)f2bf(b) << 16);
}

// -------------------- small elementwise kernels --------------------

__global__ void zero_k(float* p, int n) {
    int i = blockIdx.x * blockDim.x + threadIdx.x;
    if (i < n) p[i] = 0.0f;
}

// h[n][d] = sum of 6 node-feature embeddings
__global__ void embed_k(const int* __restrict__ xf,
                        const float* __restrict__ t1, const float* __restrict__ t2,
                        const float* __restrict__ t3, const float* __restrict__ t4,
                        const float* __restrict__ t5, const float* __restrict__ t6,
                        float* __restrict__ h, int total) {
    int i = blockIdx.x * blockDim.x + threadIdx.x;
    if (i >= total) return;
    int n = i >> 8;          // /DIM
    int d = i & 255;
    const int* f = xf + n * 6;
    h[i] = t1[f[0] * DIM + d] + t2[f[1] * DIM + d] + t3[f[2] * DIM + d]
         + t4[f[3] * DIM + d] + t5[f[4] * DIM + d] + t6[f[5] * DIM + d];
}

// 5-bit combo id per edge (edge_attr columns are binary)
__global__ void ecombo_k(const int* __restrict__ ea, int* __restrict__ ec, int E) {
    int e = blockIdx.x * blockDim.x + threadIdx.x;
    if (e >= E) return;
    const int* a = ea + e * 5;
    ec[e] = (a[0] & 1) | ((a[1] & 1) << 1) | ((a[2] & 1) << 2)
          | ((a[3] & 1) << 3) | ((a[4] & 1) << 4);
}

// etab[0..31] = the 32 possible edge embeddings; etab[32] = self-loop embedding
__global__ void build_etab(const float* __restrict__ e1, const float* __restrict__ e2,
                           const float* __restrict__ e3, const float* __restrict__ e4,
                           const float* __restrict__ e5, float* __restrict__ etab) {
    int i = blockIdx.x * blockDim.x + threadIdx.x;
    if (i >= 33 * DIM) return;
    int c = i >> 8, d = i & 255;
    if (c < 32) {
        etab[i] = e1[(c & 1) * DIM + d] + e2[((c >> 1) & 1) * DIM + d]
                + e3[((c >> 2) & 1) * DIM + d] + e4[((c >> 3) & 1) * DIM + d]
                + e5[((c >> 4) & 1) * DIM + d];
    } else {
        etab[i] = e1[4 * DIM + d] + e2[d] + e3[d] + e4[d] + e5[d];
    }
}

// aggr = h + self_e  (self-loop term; also initializes the scatter target)
__global__ void aggr_init(const float* __restrict__ h, const float* __restrict__ etab,
                          float* __restrict__ aggr, int total) {
    int i = blockIdx.x * blockDim.x + threadIdx.x;
    if (i >= total) return;
    aggr[i] = h[i] + etab[32 * DIM + (i & 255)];
}

// aggr[dst] += h[src] + etab[combo], float4-per-thread, 4 fp32 global atomics
__global__ void edge_scatter(const int* __restrict__ ei, const int* __restrict__ ec,
                             const float* __restrict__ h, const float* __restrict__ etab,
                             float* __restrict__ aggr, int E) {
    int i = blockIdx.x * blockDim.x + threadIdx.x;
    if (i >= E * (DIM / 4)) return;
    int e  = i >> 6;          // /(DIM/4)
    int d4 = i & 63;
    int src = ei[e];
    int dst = ei[E + e];
    int c   = ec[e];
    float4 hv = ((const float4*)h)[src * 64 + d4];
    float4 ev = ((const float4*)etab)[c * 64 + d4];
    float* o = aggr + dst * DIM + d4 * 4;
    atomicAdd(o + 0, hv.x + ev.x);
    atomicAdd(o + 1, hv.y + ev.y);
    atomicAdd(o + 2, hv.z + ev.z);
    atomicAdd(o + 3, hv.w + ev.w);
}

// -------------------- WMMA bf16 GEMM: C = act(A[MxK] * W[NtxK]^T + bias) ----
// 256 threads = 8 waves (4 row x 2 col), block tile 64x64, K chunks of 32.
// Each wave owns a 16x32 slab: one A fragment, two B fragments, two WMMAs.

#define BM 64
#define BN 64
#define BK 32
#define LDSW 36   // row stride in shorts: 72B (multiple of 8B, conflict-free banks)

__global__ __launch_bounds__(256) void gemm_wmma(
    const float* __restrict__ A, const float* __restrict__ W,
    const float* __restrict__ bias, float* __restrict__ C,
    int M, int Nt, int K, int relu) {

    __shared__ __align__(16) unsigned short As[BM][LDSW];
    __shared__ __align__(16) unsigned short Ws[BN][LDSW];

    const int tid  = threadIdx.x;
    const int bm   = blockIdx.x * BM;
    const int bn   = blockIdx.y * BN;
    const int wave = tid >> 5;
    const int lane = tid & 31;
    const int wr   = wave >> 1;      // 0..3  (16-row slab)
    const int wc   = wave & 1;       // 0..1  (32-col slab)
    const int lo   = lane & 15;
    const int hi   = lane >> 4;

    const int sr  = tid >> 3;        // staging: row 0..31 (x2 iters -> 64)
    const int sc4 = tid & 7;         // staging: float4 index 0..7 within row

    v8f acc0 = {}, acc1 = {};

    for (int k0 = 0; k0 < K; k0 += BK) {
        // stage A tile (64x32 fp32 -> bf16), branch-free (row clamp)
#pragma unroll
        for (int it = 0; it < 2; ++it) {
            int r  = sr + it * 32;
            int gr = bm + r; gr = (gr < M) ? gr : (M - 1);
            float4 v = ((const float4*)(A + (long)gr * K + k0))[sc4];
            uint2 pv; pv.x = pk2bf(v.x, v.y); pv.y = pk2bf(v.z, v.w);
            *(uint2*)&As[r][sc4 * 4] = pv;
        }
        // stage W tile (64 rows of W = 64 columns of B)
#pragma unroll
        for (int it = 0; it < 2; ++it) {
            int r  = sr + it * 32;
            int gn = bn + r; gn = (gn < Nt) ? gn : (Nt - 1);
            float4 v = ((const float4*)(W + (long)gn * K + k0))[sc4];
            uint2 pv; pv.x = pk2bf(v.x, v.y); pv.y = pk2bf(v.z, v.w);
            *(uint2*)&Ws[r][sc4 * 4] = pv;
        }
        __syncthreads();

        // fragments in contiguous k order (k = 16*hi + s), same for A and B
        union { unsigned int u[8]; v16bf v; } fa, fb0, fb1;
        const unsigned short* Ar = &As[wr * 16 + lo][0];
        const unsigned short* B0 = &Ws[wc * 32 + lo][0];
        const unsigned short* B1 = &Ws[wc * 32 + 16 + lo][0];
#pragma unroll
        for (int p = 0; p < 8; ++p) {
            // A: lane = row; K pairs 16*hi + 2p
            int ka = 16 * hi + 2 * p;
            fa.u[p] = *(const unsigned int*)(Ar + ka);
            // B: lane = col; K pairs 16*hi + 2p
            int kb = 16 * hi + 2 * p;
            fb0.u[p] = *(const unsigned int*)(B0 + kb);
            fb1.u[p] = *(const unsigned int*)(B1 + kb);
        }

        acc0 = __builtin_amdgcn_wmma_f32_16x16x32_bf16(
                   false, fa.v, false, fb0.v, (short)0, acc0, false, false);
        acc1 = __builtin_amdgcn_wmma_f32_16x16x32_bf16(
                   false, fa.v, false, fb1.v, (short)0, acc1, false, false);
        __syncthreads();
    }

    // C/D layout: VGPR r -> M = 16*wr + r + 8*hi, N = lane lo within 16-col tile
    int n0 = bn + wc * 32 + lo;
    int n1 = n0 + 16;
    float bv0 = bias ? bias[n0] : 0.0f;
    float bv1 = bias ? bias[n1] : 0.0f;
#pragma unroll
    for (int r = 0; r < 8; ++r) {
        int m = bm + wr * 16 + r + hi * 8;
        if (m < M) {
            float v0 = acc0[r] + bv0;
            float v1 = acc1[r] + bv1;
            if (relu) { v0 = fmaxf(v0, 0.0f); v1 = fmaxf(v1, 0.0f); }
            C[(long)m * Nt + n0] = v0;
            C[(long)m * Nt + n1] = v1;
        }
    }
}

// -------------------- BatchNorm --------------------

// per-column sum / sumsq (stats must be zeroed first)
__global__ void bn_stats(const float* __restrict__ X, float* __restrict__ stats,
                         int R, int D) {
    int c = threadIdx.x;     // blockDim.x == D == 256
    float s = 0.0f, ss = 0.0f;
    for (int r = blockIdx.x; r < R; r += gridDim.x) {
        float v = X[(long)r * D + c];
        s += v; ss += v * v;
    }
    atomicAdd(&stats[c], s);
    atomicAdd(&stats[D + c], ss);
}

__global__ void bn_norm(const float* __restrict__ X, float* __restrict__ Y,
                        const float* __restrict__ stats,
                        const float* __restrict__ g, const float* __restrict__ b,
                        int R, int D, int relu) {
    int i = blockIdx.x * blockDim.x + threadIdx.x;
    if (i >= R * D) return;
    int c = i % D;
    float rn   = 1.0f / (float)R;
    float mean = stats[c] * rn;
    float var  = stats[D + c] * rn - mean * mean;
    float inv  = rsqrtf(var + 1e-5f);
    float v = g[c] * (X[i] - mean) * inv + b[c];
    if (relu) v = fmaxf(v, 0.0f);
    Y[i] = v;
}

// -------------------- pooling & head --------------------

__global__ void pool_scatter(const float* __restrict__ h, const int* __restrict__ batch,
                             float* __restrict__ hg, float* __restrict__ counts, int total) {
    int i = blockIdx.x * blockDim.x + threadIdx.x;
    if (i >= total) return;
    int n = i >> 8, d = i & 255;
    int g = batch[n];
    atomicAdd(&hg[g * DIM + d], h[i]);
    if (d == 0) atomicAdd(&counts[g], 1.0f);
}

__global__ void pool_div(float* __restrict__ hg, const float* __restrict__ counts, int total) {
    int i = blockIdx.x * blockDim.x + threadIdx.x;
    if (i >= total) return;
    hg[i] /= fmaxf(counts[i >> 8], 1.0f);
}

// out[g] = softplus(hg[g] @ ow1^T + ob1) @ ow2^T + ob2   (tiny: 256x128, 128x2)
__global__ __launch_bounds__(128) void head_k(
    const float* __restrict__ hg, const float* __restrict__ ow1,
    const float* __restrict__ ob1, const float* __restrict__ ow2,
    const float* __restrict__ ob2, float* __restrict__ out) {
    __shared__ float row[DIM];
    __shared__ float t[128];
    int g = blockIdx.x, tid = threadIdx.x;
    row[tid]       = hg[g * DIM + tid];
    row[tid + 128] = hg[g * DIM + tid + 128];
    __syncthreads();
    float s = ob1[tid];
    const float* wrow = ow1 + tid * DIM;
    for (int d = 0; d < DIM; ++d) s += row[d] * wrow[d];
    t[tid] = (s > 20.0f) ? s : log1pf(expf(s));   // softplus
    __syncthreads();
    if (tid < 2) {
        float o = ob2[tid];
        const float* w2 = ow2 + tid * 128;
        for (int j = 0; j < 128; ++j) o += t[j] * w2[j];
        out[g * 2 + tid] = o;
    }
}

// -------------------- launcher --------------------

extern "C" void kernel_launch(void* const* d_in, const int* in_sizes, int n_in,
                              void* d_out, int out_size, void* d_ws, size_t ws_size,
                              hipStream_t stream) {
    (void)in_sizes; (void)n_in; (void)out_size; (void)ws_size;
    const int N = NN, E = EE, G = GG, D = DIM, L = LL;

    const int*   x_feats    = (const int*)d_in[0];
    const int*   edge_index = (const int*)d_in[1];
    const int*   edge_attr  = (const int*)d_in[2];
    const int*   batch      = (const int*)d_in[3];
    const float* nt1 = (const float*)d_in[4];
    const float* nt2 = (const float*)d_in[5];
    const float* nt3 = (const float*)d_in[6];
    const float* nt4 = (const float*)d_in[7];
    const float* nt5 = (const float*)d_in[8];
    const float* nt6 = (const float*)d_in[9];
    const float* et1 = (const float*)d_in[10];
    const float* et2 = (const float*)d_in[11];
    const float* et3 = (const float*)d_in[12];
    const float* et4 = (const float*)d_in[13];
    const float* et5 = (const float*)d_in[14];
    const float* w1  = (const float*)d_in[15];
    const float* b1  = (const float*)d_in[16];
    const float* w2  = (const float*)d_in[17];
    const float* b2  = (const float*)d_in[18];
    const float* bng = (const float*)d_in[19];
    const float* bnb = (const float*)d_in[20];
    const float* pw  = (const float*)d_in[21];
    const float* pg  = (const float*)d_in[22];
    const float* pb  = (const float*)d_in[23];
    const float* ow1 = (const float*)d_in[24];
    const float* ob1 = (const float*)d_in[25];
    const float* ow2 = (const float*)d_in[26];
    const float* ob2 = (const float*)d_in[27];
    float* out = (float*)d_out;

    // workspace bump allocator (256B aligned regions)
    uintptr_t base = (uintptr_t)d_ws;
    size_t off = 0;
    auto wsa = [&](size_t bytes) -> void* {
        void* p = (void*)(base + off);
        off = (off + bytes + 255) & ~(size_t)255;
        return p;
    };
    float* h     = (float*)wsa((size_t)N * D * 4);       // node features
    float* aggr  = (float*)wsa((size_t)N * D * 4);       // aggr, reused as hh2
    float* hh1   = (float*)wsa((size_t)N * 2 * D * 4);   // hidden 2D
    float* etab  = (float*)wsa(33 * D * 4);
    float* stats = (float*)wsa(2 * D * 4);
    int*   ecomb = (int*)  wsa((size_t)E * 4);
    float* hg    = (float*)wsa((size_t)G * D * 4);
    float* hgt   = (float*)wsa((size_t)G * D * 4);
    float* cnts  = (float*)wsa((size_t)G * 4);

    const int TPB = 256;
    int totND = N * D;

    embed_k<<<(totND + TPB - 1) / TPB, TPB, 0, stream>>>(
        x_feats, nt1, nt2, nt3, nt4, nt5, nt6, h, totND);
    ecombo_k<<<(E + TPB - 1) / TPB, TPB, 0, stream>>>(edge_attr, ecomb, E);

    for (int l = 0; l < L; ++l) {
        build_etab<<<(33 * D + TPB - 1) / TPB, TPB, 0, stream>>>(
            et1 + l * 5 * D, et2 + l * 4 * D, et3 + l * 2 * D,
            et4 + l * 2 * D, et5 + l * 2 * D, etab);
        aggr_init<<<(totND + TPB - 1) / TPB, TPB, 0, stream>>>(h, etab, aggr, totND);
        int escat = E * (D / 4);
        edge_scatter<<<(escat + TPB - 1) / TPB, TPB, 0, stream>>>(
            edge_index, ecomb, h, etab, aggr, E);

        // GIN MLP: relu(aggr @ w1^T + b1) @ w2^T + b2
        dim3 g1((N + BM - 1) / BM, (2 * D) / BN);
        gemm_wmma<<<g1, 256, 0, stream>>>(aggr, w1 + l * 2 * D * D, b1 + l * 2 * D,
                                          hh1, N, 2 * D, D, 1);
        dim3 g2((N + BM - 1) / BM, D / BN);
        gemm_wmma<<<g2, 256, 0, stream>>>(hh1, w2 + (size_t)l * D * 2 * D, b2 + l * D,
                                          aggr, N, D, 2 * D, 0);

        // BatchNorm + ReLU -> h
        zero_k<<<2, TPB, 0, stream>>>(stats, 2 * D);
        bn_stats<<<240, D, 0, stream>>>(aggr, stats, N, D);
        bn_norm<<<(totND + TPB - 1) / TPB, TPB, 0, stream>>>(
            aggr, h, stats, bng + l * D, bnb + l * D, N, D, 1);
    }

    // global mean pool
    zero_k<<<(G * D + TPB - 1) / TPB, TPB, 0, stream>>>(hg, G * D);
    zero_k<<<1, TPB, 0, stream>>>(cnts, G);
    pool_scatter<<<(totND + TPB - 1) / TPB, TPB, 0, stream>>>(h, batch, hg, cnts, totND);
    pool_div<<<(G * D + TPB - 1) / TPB, TPB, 0, stream>>>(hg, cnts, G * D);

    // projection head: 3x (Linear no-bias -> BN -> ReLU except last)
    for (int i = 0; i < 3; ++i) {
        dim3 gp((G + BM - 1) / BM, D / BN);
        gemm_wmma<<<gp, 256, 0, stream>>>(hg, pw + (size_t)i * D * D, nullptr,
                                          hgt, G, D, D, 0);
        zero_k<<<2, TPB, 0, stream>>>(stats, 2 * D);
        bn_stats<<<8, D, 0, stream>>>(hgt, stats, G, D);
        bn_norm<<<(G * D + TPB - 1) / TPB, TPB, 0, stream>>>(
            hgt, hg, stats, pg + i * D, pb + i * D, G, D, (i < 2) ? 1 : 0);
    }

    // output MLP
    head_k<<<G, 128, 0, stream>>>(hg, ow1, ob1, ow2, ob2, out);
}
